// MultiHeadAttention_36386962932374
// MI455X (gfx1250) — matrix operand order repaired and checked by reading, not compiled, not run on hardware
//
#include <hip/hip_runtime.h>

// Problem constants (match the reference)
#define T_SEQ 1024
#define BB    8
#define EE    768
#define HH    12
#define DD    64
#define SCALING 0.125f   // D^-0.5

typedef __bf16 bf16_t;
typedef __attribute__((ext_vector_type(16))) __bf16 v16bf;
typedef __attribute__((ext_vector_type(8)))  __bf16 v8bf;
typedef __attribute__((ext_vector_type(8)))  float  v8f;

__device__ __forceinline__ v16bf combine16(v8bf lo, v8bf hi) {
    return __builtin_shufflevector(lo, hi, 0,1,2,3,4,5,6,7,8,9,10,11,12,13,14,15);
}

__device__ __forceinline__ v8f wmma_bf16(v16bf a, v16bf b, v8f c) {
    // D = A(16x32 bf16) * B(32x16 bf16) + C(16x16 f32)
    return __builtin_amdgcn_wmma_f32_16x16x32_bf16(false, a, false, b,
                                                   (short)0, c, false, false);
}

// ---------------------------------------------------------------------------
// Kernel 0: one-time f32 -> bf16 conversion of a weight matrix (E*E elems).
// Removes all v_cvt_pk_bf16_f32 from the GEMM inner loops.
// ---------------------------------------------------------------------------
__global__ __launch_bounds__(256) void cvt_w_kernel(
    const float* __restrict__ src, bf16_t* __restrict__ dst)
{
    const int i = (blockIdx.x * 256 + threadIdx.x) * 8;   // E*E = 589824, /8 per thread
    const v8f x = *(const v8f*)(src + i);
    v8bf y;
#pragma unroll
    for (int j = 0; j < 8; ++j) y[j] = (bf16_t)x[j];
    *(v8bf*)(dst + i) = y;
}

// ---------------------------------------------------------------------------
// Kernel 1: QKV projection.  X[8192,768] @ W^T + b  for Wq/Wk/Wv (bf16 W).
//   q -> [B,H,T,D] bf16 (scaled), k -> [B,H,T,D] bf16, v -> [B,H,D,T] bf16
// ---------------------------------------------------------------------------
#define XSTR 776   // 768 + 8 pad (bf16 elems) to break LDS bank conflicts

__global__ __launch_bounds__(256) void qkv_proj_kernel(
    const float* __restrict__ x,                      // [T*B, E] f32
    const bf16_t* __restrict__ Wq, const float* __restrict__ bq,
    const bf16_t* __restrict__ Wk, const float* __restrict__ bk,
    const bf16_t* __restrict__ Wv, const float* __restrict__ bv,
    bf16_t* __restrict__ q_ws,                        // [B,H,T,D]
    bf16_t* __restrict__ k_ws,                        // [B,H,T,D]
    bf16_t* __restrict__ vT_ws)                       // [B,H,D,T]
{
    __shared__ bf16_t xt[16 * XSTR];

    const int m0   = blockIdx.x * 16;
    const int tid  = threadIdx.x;
    const int wave = tid >> 5;
    const int lane = tid & 31;
    const int lrow = lane & 15;
    const int lhi  = lane >> 4;

    // Stage the 16x768 input tile as bf16 in LDS (float4 loads, 8B LDS stores).
    for (int idx = tid; idx < 16 * (EE / 4); idx += 256) {
        const int r = idx / (EE / 4), e4 = idx % (EE / 4);
        const float4 xv = ((const float4*)(x + (size_t)(m0 + r) * EE))[e4];
        bf16_t* p = &xt[r * XSTR + e4 * 4];
        p[0] = (bf16_t)xv.x; p[1] = (bf16_t)xv.y;
        p[2] = (bf16_t)xv.z; p[3] = (bf16_t)xv.w;
    }
    __syncthreads();

    // 144 output tiles: j/48 selects q|k|v, (j%48)*16 is the N offset.
    for (int j = wave; j < 144; j += 8) {
        const int mat = j / 48;
        const int n0  = (j % 48) * 16;
        const bf16_t* W  = (mat == 0) ? Wq : (mat == 1) ? Wk : Wv;
        const float*  bi = (mat == 0) ? bq : (mat == 1) ? bk : bv;
        const int n = n0 + lrow;

        v8f c;
        const float b0 = bi[n];
#pragma unroll
        for (int r = 0; r < 8; ++r) c[r] = b0;   // bias as initial accumulator

#pragma unroll 4
        for (int kk = 0; kk < EE / 32; ++kk) {
            // A fragment from LDS: row = lrow, K contiguous 16*lhi..16*lhi+15
            const bf16_t* ap = &xt[lrow * XSTR + kk * 32 + lhi * 16];
            const v16bf a = combine16(*(const v8bf*)ap, *(const v8bf*)(ap + 8));
            // B fragment: column n = W row (contiguous), already bf16
            const v16bf b = *(const v16bf*)&W[(size_t)n * EE + kk * 32 + lhi * 16];
            c = wmma_bf16(a, b, c);
        }
        if (mat == 0) {
#pragma unroll
            for (int r = 0; r < 8; ++r) c[r] *= SCALING;
        }

        const int h = n / DD, d = n % DD;
#pragma unroll
        for (int r = 0; r < 8; ++r) {
            const int row = m0 + r + 8 * lhi;     // flattened t*B + b
            const int t = row >> 3;               // / B
            const int b = row & 7;                // % B
            const size_t bh = (size_t)b * HH + h;
            if (mat == 2)
                vT_ws[(bh * DD + d) * T_SEQ + t] = (bf16_t)c[r];
            else if (mat == 0)
                q_ws[(bh * T_SEQ + t) * DD + d] = (bf16_t)c[r];
            else
                k_ws[(bh * T_SEQ + t) * DD + d] = (bf16_t)c[r];
        }
    }
}

// ---------------------------------------------------------------------------
// Kernel 2: attention core. One WG per (b, 16-query-row tile); loops heads.
//   scores = q @ k^T + bias (bias streamed NT as WMMA accumulator)
//   exact softmax over full row (T=1024) in LDS, probs -> bf16
//   attn = probs @ v (8-wave K-split + LDS reduce)
//   mean-over-heads of probs accumulated in LDS, written NT at the end
// ---------------------------------------------------------------------------
#define QSTR 72    // q tile LDS row stride (bf16)
#define SSTR 1025  // scores/mean LDS row stride (f32)
#define PSTR 1032  // probs LDS row stride (bf16), multiple of 8

#define OFF_Q     0
#define OFF_SC    (OFF_Q  + 16 * QSTR * 2)        // 2304
#define OFF_PR    (OFF_SC + 16 * SSTR * 4)        // 67904
#define OFF_MEAN  (OFF_PR + 16 * PSTR * 2)        // 100928
#define OFF_RED   (OFF_MEAN + 16 * SSTR * 4)      // 166528
#define OFF_RMAX  (OFF_RED + 16 * 16 * 4)         // 167552
#define OFF_RSUM  (OFF_RMAX + 16 * 4)             // 167616
#define SMEM2_BYTES (OFF_RSUM + 16 * 4)           // 167680 (< 320 KB WGP LDS)

__global__ __launch_bounds__(256) void attn_kernel(
    const float*  __restrict__ attn_bias,   // [B,H,T,T]
    const bf16_t* __restrict__ q_ws,        // [B,H,T,D]
    const bf16_t* __restrict__ k_ws,        // [B,H,T,D]
    const bf16_t* __restrict__ vT_ws,       // [B,H,D,T]
    bf16_t* __restrict__ attn_ws,           // [T*B, E]
    float*  __restrict__ mean_out)          // [B,T,T]
{
    extern __shared__ char smem[];
    bf16_t* q_lds   = (bf16_t*)(smem + OFF_Q);
    float*  scores  = (float*) (smem + OFF_SC);
    bf16_t* probs   = (bf16_t*)(smem + OFF_PR);
    float*  meanbuf = (float*) (smem + OFF_MEAN);
    float*  red     = (float*) (smem + OFF_RED);
    float*  rowmax  = (float*) (smem + OFF_RMAX);
    float*  rowsum  = (float*) (smem + OFF_RSUM);

    const int t0   = blockIdx.x * 16;
    const int b    = blockIdx.y;
    const int tid  = threadIdx.x;
    const int wave = tid >> 5;
    const int lane = tid & 31;
    const int lrow = lane & 15;
    const int lhi  = lane >> 4;
    const int r16  = tid & 15;      // softmax: score row
    const int sg   = tid >> 4;      // softmax: 64-wide segment

    for (int idx = tid; idx < 16 * SSTR; idx += 256) meanbuf[idx] = 0.0f;
    __syncthreads();

    for (int h = 0; h < HH; ++h) {
        const size_t bh = (size_t)b * HH + h;

        // q tile -> LDS
        for (int idx = tid; idx < 16 * DD; idx += 256) {
            const int r = idx >> 6, d = idx & 63;
            q_lds[r * QSTR + d] = q_ws[(bh * T_SEQ + t0 + r) * DD + d];
        }
        __syncthreads();

        // ---- scores: 64 N-tiles of 16, bias streamed (NT) as accumulator ----
        const float* biasp = attn_bias + (bh * T_SEQ + t0) * (size_t)T_SEQ;
        for (int nt = wave; nt < 64; nt += 8) {
            const int s0 = nt * 16;
            v8f c;
#pragma unroll
            for (int r = 0; r < 8; ++r)
                c[r] = __builtin_nontemporal_load(
                           &biasp[(size_t)(r + 8 * lhi) * T_SEQ + s0 + lrow]);
#pragma unroll
            for (int kk = 0; kk < 2; ++kk) {
                const bf16_t* ap = &q_lds[lrow * QSTR + kk * 32 + lhi * 16];
                const v16bf a = combine16(*(const v8bf*)ap, *(const v8bf*)(ap + 8));
                // B: column s = k row (contiguous in d)
                const v16bf bb = *(const v16bf*)
                    &k_ws[(bh * T_SEQ + s0 + lrow) * DD + kk * 32 + lhi * 16];
                c = wmma_bf16(a, bb, c);
            }
#pragma unroll
            for (int r = 0; r < 8; ++r)
                scores[(r + 8 * lhi) * SSTR + s0 + lrow] = c[r];
        }
        __syncthreads();

        // ---- exact softmax over the full 1024-wide row ----
        {
            float m = -3.4e38f;
            const float* srow = &scores[r16 * SSTR + sg * 64];
#pragma unroll 8
            for (int i = 0; i < 64; ++i) m = fmaxf(m, srow[i]);
            red[sg * 16 + r16] = m;
        }
        __syncthreads();
        if (sg == 0) {
            float m = red[r16];
#pragma unroll
            for (int j2 = 1; j2 < 16; ++j2) m = fmaxf(m, red[j2 * 16 + r16]);
            rowmax[r16] = m;
        }
        __syncthreads();
        {
            const float m = rowmax[r16];
            float s = 0.0f;
            float* srow = &scores[r16 * SSTR + sg * 64];
#pragma unroll 8
            for (int i = 0; i < 64; ++i) {
                const float p = __expf(srow[i] - m);
                srow[i] = p;
                s += p;
            }
            red[sg * 16 + r16] = s;
        }
        __syncthreads();
        if (sg == 0) {
            float s = 0.0f;
#pragma unroll
            for (int j2 = 0; j2 < 16; ++j2) s += red[j2 * 16 + r16];
            rowsum[r16] = s;
        }
        __syncthreads();
        {
            const float inv = 1.0f / rowsum[r16];
            const float* srow = &scores[r16 * SSTR + sg * 64];
            bf16_t* prow = &probs[r16 * PSTR + sg * 64];
            float*  mrow = &meanbuf[r16 * SSTR + sg * 64];
#pragma unroll 8
            for (int i = 0; i < 64; ++i) {
                const float p = srow[i] * inv;
                prow[i] = (bf16_t)p;
                mrow[i] += p * (1.0f / (float)HH);
            }
        }
        __syncthreads();

        // ---- PV: probs[16,1024] @ v[1024,64]; 8 waves = 4 d-tiles x 2 K-halves
        {
            const int d0    = (wave & 3) * 16;
            const int khalf = wave >> 2;
            v8f c;
#pragma unroll
            for (int r = 0; r < 8; ++r) c[r] = 0.0f;
#pragma unroll 4
            for (int kk = khalf * 16; kk < khalf * 16 + 16; ++kk) {
                const bf16_t* ap = &probs[lrow * PSTR + kk * 32 + lhi * 16];
                const v16bf a = combine16(*(const v8bf*)ap, *(const v8bf*)(ap + 8));
                // B: column d = vT row (contiguous in s)
                const v16bf bb = *(const v16bf*)
                    &vT_ws[(bh * DD + d0 + lrow) * T_SEQ + kk * 32 + lhi * 16];
                c = wmma_bf16(a, bb, c);
            }
            float* scratch = scores;   // scores are dead now; reuse 4 KB of it
            if (khalf == 1) {
#pragma unroll
                for (int r = 0; r < 8; ++r)
                    scratch[((wave & 3) * 16 + r + 8 * lhi) * 16 + lrow] = c[r];
            }
            __syncthreads();
            if (khalf == 0) {
#pragma unroll
                for (int r = 0; r < 8; ++r) {
                    c[r] += scratch[((wave & 3) * 16 + r + 8 * lhi) * 16 + lrow];
                    const int t = t0 + r + 8 * lhi;
                    attn_ws[((size_t)t * BB + b) * EE + h * DD + d0 + lrow] = (bf16_t)c[r];
                }
            }
        }
        __syncthreads();
    }

    // mean over heads -> second output region (streamed, no reuse -> NT)
    for (int idx = tid; idx < 16 * T_SEQ; idx += 256) {
        const int r = idx >> 10, s = idx & 1023;
        __builtin_nontemporal_store(
            meanbuf[r * SSTR + s],
            &mean_out[((size_t)b * T_SEQ + t0 + r) * T_SEQ + s]);
    }
}

// ---------------------------------------------------------------------------
// Kernel 3: output projection.  attn[8192,768](bf16) @ Wo^T + bo -> f32 out.
// attn row-tile staged in LDS; B = pre-converted bf16 Wo rows.
// ---------------------------------------------------------------------------
__global__ __launch_bounds__(256) void out_proj_kernel(
    const bf16_t* __restrict__ attn_ws,   // [T*B, E]
    const bf16_t* __restrict__ Wo,        // [E, E] bf16
    const float*  __restrict__ bo,
    float* __restrict__ out)              // [T*B, E]
{
    __shared__ bf16_t at[16 * XSTR];

    const int m0   = blockIdx.x * 16;
    const int tid  = threadIdx.x;
    const int wave = tid >> 5;
    const int lane = tid & 31;
    const int lrow = lane & 15;
    const int lhi  = lane >> 4;

    // Stage the 16x768 bf16 attn tile in LDS (16B vector copies).
    for (int idx = tid; idx < 16 * (EE / 8); idx += 256) {
        const int r = idx / (EE / 8), e8 = idx % (EE / 8);
        *(v8bf*)&at[r * XSTR + e8 * 8] =
            *(const v8bf*)&attn_ws[(size_t)(m0 + r) * EE + e8 * 8];
    }
    __syncthreads();

    for (int nt = wave; nt < 48; nt += 8) {
        const int n0 = nt * 16;
        const int n  = n0 + lrow;
        v8f c;
        const float b0 = bo[n];
#pragma unroll
        for (int r = 0; r < 8; ++r) c[r] = b0;

#pragma unroll 4
        for (int kk = 0; kk < EE / 32; ++kk) {
            const bf16_t* ap = &at[lrow * XSTR + kk * 32 + lhi * 16];
            const v16bf a = combine16(*(const v8bf*)ap, *(const v8bf*)(ap + 8));
            const v16bf b = *(const v16bf*)&Wo[(size_t)n * EE + kk * 32 + lhi * 16];
            c = wmma_bf16(a, b, c);
        }
#pragma unroll
        for (int r = 0; r < 8; ++r)
            out[(size_t)(m0 + r + 8 * lhi) * EE + n] = c[r];
    }
}

// ---------------------------------------------------------------------------
extern "C" void kernel_launch(void* const* d_in, const int* in_sizes, int n_in,
                              void* d_out, int out_size, void* d_ws, size_t ws_size,
                              hipStream_t stream) {
    (void)in_sizes; (void)n_in; (void)out_size;

    const float* query     = (const float*)d_in[0];
    const float* attn_bias = (const float*)d_in[1];
    const float* Wq = (const float*)d_in[2];
    const float* bq = (const float*)d_in[3];
    const float* Wk = (const float*)d_in[4];
    const float* bk = (const float*)d_in[5];
    const float* Wv = (const float*)d_in[6];
    const float* bv = (const float*)d_in[7];
    const float* Wo = (const float*)d_in[8];
    const float* bo = (const float*)d_in[9];

    float* out      = (float*)d_out;                         // [T,B,E]
    float* mean_out = out + (size_t)T_SEQ * BB * EE;         // [B,T,T]

    // Workspace layout (bf16): q, k, vT (each B*H*T*D), attn (T*B*E),
    // then 4 pre-converted weight matrices (E*E each).
    const size_t qkv_elems = (size_t)BB * HH * T_SEQ * DD;   // 6,291,456
    const size_t w_elems   = (size_t)EE * EE;                // 589,824
    char* ws = (char*)d_ws;
    bf16_t* q_ws    = (bf16_t*)(ws + 0 * qkv_elems * sizeof(bf16_t));
    bf16_t* k_ws    = (bf16_t*)(ws + 1 * qkv_elems * sizeof(bf16_t));
    bf16_t* vT_ws   = (bf16_t*)(ws + 2 * qkv_elems * sizeof(bf16_t));
    bf16_t* attn_ws = (bf16_t*)(ws + 3 * qkv_elems * sizeof(bf16_t));
    bf16_t* wq_bf   = attn_ws + qkv_elems;                   // attn == qkv size
    bf16_t* wk_bf   = wq_bf + w_elems;
    bf16_t* wv_bf   = wk_bf + w_elems;
    bf16_t* wo_bf   = wv_bf + w_elems;
    const size_t need = (4 * qkv_elems + 4 * w_elems) * sizeof(bf16_t); // ~55 MB
    if (ws_size < need) return;

    const int cvt_blocks = (int)(w_elems / (256 * 8));       // 288
    cvt_w_kernel<<<cvt_blocks, 256, 0, stream>>>(Wq, wq_bf);
    cvt_w_kernel<<<cvt_blocks, 256, 0, stream>>>(Wk, wk_bf);
    cvt_w_kernel<<<cvt_blocks, 256, 0, stream>>>(Wv, wv_bf);
    cvt_w_kernel<<<cvt_blocks, 256, 0, stream>>>(Wo, wo_bf);

    qkv_proj_kernel<<<(T_SEQ * BB) / 16, 256, 0, stream>>>(
        query, wq_bf, bq, wk_bf, bk, wv_bf, bv, q_ws, k_ws, vT_ws);

    dim3 g2(T_SEQ / 16, BB);
    attn_kernel<<<g2, 256, SMEM2_BYTES, stream>>>(
        attn_bias, q_ws, k_ws, vT_ws, attn_ws, mean_out);

    out_proj_kernel<<<(T_SEQ * BB) / 16, 256, 0, stream>>>(
        attn_ws, wo_bf, bo, out);
}
